// OptimalTransportSolver_49589692399737
// MI455X (gfx1250) — matrix operand order repaired and checked by reading, not compiled, not run on hardware
//
#include <hip/hip_runtime.h>
#include <hip/hip_bf16.h>

// ---------------------------------------------------------------------------
// Sinkhorn OT on MI455X (gfx1250):
//   K = exp(-C/(max(C)*reg)) held in L2 as bf16 (32MB) + explicit K^T (32MB).
//   100 iterations of  a = p/(K b);  b = q/(K^T a)  as WMMA bf16 matvecs.
//   Final plan computed in f32 from original C for full output precision.
// Deterministic: no float atomics anywhere (uint atomicMax is order-free,
// cross-wave reduction is fixed-order via LDS slots).
// ---------------------------------------------------------------------------

#define OT_N 4096
#define OT_REG 0.05f

typedef __attribute__((ext_vector_type(16))) __bf16 v16bf;
typedef __attribute__((ext_vector_type(8)))  __bf16 v8bf;
typedef __attribute__((ext_vector_type(8)))  float  v8f;

// --- init: cmax accumulator = 0, b = ones (v0 = 0 -> b0 = exp(0) = 1) ------
__global__ void ot_init(unsigned* cmax_bits, float* b_f, __bf16* b_b) {
    int i = blockIdx.x * blockDim.x + threadIdx.x;
    if (i == 0) *cmax_bits = 0u;
    if (i < OT_N) { b_f[i] = 1.0f; b_b[i] = (__bf16)1.0f; }
}

// --- max(C): C is uniform[0,1) >= 0, so float bits compare as uint ---------
__global__ __launch_bounds__(256) void ot_maxc(const unsigned* __restrict__ Cb,
                                               unsigned* cmax_bits, int n) {
    __shared__ unsigned sm[256];
    unsigned m = 0u;
    for (int i = blockIdx.x * blockDim.x + threadIdx.x; i < n;
         i += gridDim.x * blockDim.x)
        m = max(m, Cb[i]);
    sm[threadIdx.x] = m;
    __syncthreads();
    for (int s = 128; s > 0; s >>= 1) {
        if (threadIdx.x < s) sm[threadIdx.x] = max(sm[threadIdx.x], sm[threadIdx.x + s]);
        __syncthreads();
    }
    if (threadIdx.x == 0) atomicMax(cmax_bits, sm[0]);
}

// --- K = exp(-C*s) as bf16, row-major, plus coalesced transpose K^T --------
__global__ __launch_bounds__(256) void ot_buildK(const float* __restrict__ C,
                                                 const unsigned* __restrict__ cmax_bits,
                                                 __bf16* __restrict__ K,
                                                 __bf16* __restrict__ KT) {
    __shared__ float tile[32][33];
    const float s = 1.0f / (__uint_as_float(*cmax_bits) * OT_REG);
    const int tx = threadIdx.x;           // 0..31
    const int ty = threadIdx.y;           // 0..7
    const int bx = blockIdx.x * 32, by = blockIdx.y * 32;
#pragma unroll
    for (int r = 0; r < 32; r += 8) {
        const int row = by + ty + r, col = bx + tx;
        const float e = __expf(-C[(long)row * OT_N + col] * s);
        K[(long)row * OT_N + col] = (__bf16)e;
        tile[ty + r][tx] = e;
    }
    __syncthreads();
#pragma unroll
    for (int r = 0; r < 32; r += 8) {
        const int row = bx + ty + r, col = by + tx;
        KT[(long)row * OT_N + col] = (__bf16)tile[tx][ty + r];
    }
}

// --- core: out = w / (Kmat * xin), one 16-row band per block ---------------
// 16 waves split the K-dim into 256-wide slices; each slice = 8x
// v_wmma_f32_16x16x32_bf16 with the vector replicated across B's columns.
#define MV_WAVES  16
#define MV_TPB    (MV_WAVES * 32)
#define MV_SLICE  (OT_N / MV_WAVES)     // 256
#define MV_CHUNKS (MV_SLICE / 32)       // 8

__global__ __launch_bounds__(MV_TPB) void ot_matvec_div(const __bf16* __restrict__ Kmat,
                                                        const __bf16* __restrict__ xin,
                                                        const float*  __restrict__ w,
                                                        float*  __restrict__ yout_f,
                                                        __bf16* __restrict__ yout_b) {
    __shared__ float part[MV_WAVES][16];
    const int rowBase = blockIdx.x * 16;
    const int wave    = threadIdx.x >> 5;    // 0..15 (wave32)
    const int lane    = threadIdx.x & 31;
    const int halfSel = lane >> 4;           // 0: lanes 0-15, 1: lanes 16-31
    const int mrow    = lane & 15;

    const int  kBegin = wave * MV_SLICE;
    const long rowOff = (long)(rowBase + mrow) * OT_N;

    v8f acc = {};
#pragma unroll 4
    for (int it = 0; it < MV_CHUNKS; ++it) {   // constant trip count -> scalar loop
        const int k0 = kBegin + (it << 5);
        // A (16x32 bf16): lanes 0-15 -> K {k0..k0+7, k0+16..k0+23}
        //                 lanes16-31 -> K {k0+8..k0+15, k0+24..k0+31}
        const __bf16* ap = Kmat + rowOff + k0 + (halfSel ? 8 : 0);
        const v8bf lo = *(const v8bf*)(ap);        // 16B load
        const v8bf hi = *(const v8bf*)(ap + 16);   // 16B load
        v16bf A;
#pragma unroll
        for (int i = 0; i < 8; ++i) { A[i] = lo[i]; A[8 + i] = hi[i]; }

        // B (32x16 bf16): vector chunk replicated across the 16 columns.
        // lanes 0-15 hold K {k0..k0+7, k0+16..k0+23}, lanes 16-31 hold K {k0+8..k0+15, k0+24..k0+31}.
        const __bf16* bp = xin + k0 + (halfSel ? 8 : 0);
        const v8bf blo = *(const v8bf*)(bp);
        const v8bf bhi = *(const v8bf*)(bp + 16);
        v16bf B;
#pragma unroll
        for (int i = 0; i < 8; ++i) { B[i] = blo[i]; B[8 + i] = bhi[i]; }

        acc = __builtin_amdgcn_wmma_f32_16x16x32_bf16(
            /*neg_a=*/false, A, /*neg_b=*/false, B,
            /*c_mod=*/(short)0, acc, /*reuse_a=*/false, /*reuse_b=*/false);
    }

    // Column 0 of the 16x16 accumulator: lane 0 has M=0..7 (VGPR 0..7),
    // lane 16 has M=8..15. Deterministic cross-wave reduction: each wave
    // writes its own LDS slot, then 16 threads sum in fixed wave order.
    if ((lane & 15) == 0) {
        float* dst = &part[wave][halfSel * 8];
#pragma unroll
        for (int i = 0; i < 8; ++i) dst[i] = acc[i];
    }
    __syncthreads();

    if (threadIdx.x < 16) {
        float ssum = 0.0f;
#pragma unroll
        for (int w2 = 0; w2 < MV_WAVES; ++w2) ssum += part[w2][threadIdx.x];
        const int r = rowBase + threadIdx.x;
        const float y = w[r] / ssum;
        yout_f[r] = y;
        yout_b[r] = (__bf16)y;
    }
}

// --- final plan: P = a_i * exp(-C*s) * b_j (f32, vectorized x4) ------------
__global__ __launch_bounds__(256) void ot_plan(const float* __restrict__ C,
                                               const unsigned* __restrict__ cmax_bits,
                                               const float* __restrict__ a,
                                               const float* __restrict__ b,
                                               float* __restrict__ P) {
    const float s = 1.0f / (__uint_as_float(*cmax_bits) * OT_REG);
    const long n4 = (long)OT_N * OT_N / 4;
    const long stride = (long)gridDim.x * blockDim.x;
    const float4* C4 = (const float4*)C;
    float4* P4 = (float4*)P;
    for (long i = (long)blockIdx.x * blockDim.x + threadIdx.x; i < n4; i += stride) {
        const long e0 = i * 4;
        const int row = (int)(e0 >> 12);
        const int col = (int)(e0 & (OT_N - 1));
        const float4 c = C4[i];
        const float ar = a[row];
        float4 o;
        o.x = ar * __expf(-c.x * s) * b[col + 0];
        o.y = ar * __expf(-c.y * s) * b[col + 1];
        o.z = ar * __expf(-c.z * s) * b[col + 2];
        o.w = ar * __expf(-c.w * s) * b[col + 3];
        P4[i] = o;
    }
}

extern "C" void kernel_launch(void* const* d_in, const int* in_sizes, int n_in,
                              void* d_out, int out_size, void* d_ws, size_t ws_size,
                              hipStream_t stream) {
    (void)in_sizes; (void)n_in; (void)out_size; (void)ws_size;
    const float* p = (const float*)d_in[0];
    const float* q = (const float*)d_in[1];
    const float* C = (const float*)d_in[2];
    float* out = (float*)d_out;

    // Workspace layout: [cmax:256B][K bf16:32MB][KT bf16:32MB][a_f][b_f][a_b][b_b]
    char* ws = (char*)d_ws;
    const size_t KB = (size_t)OT_N * OT_N * sizeof(__bf16);   // 32 MB
    unsigned* cmax = (unsigned*)ws;
    __bf16* K  = (__bf16*)(ws + 256);
    __bf16* KT = (__bf16*)(ws + 256 + KB);
    float*  a_f = (float*)(ws + 256 + 2 * KB);
    float*  b_f = a_f + OT_N;
    __bf16* a_b = (__bf16*)(b_f + OT_N);
    __bf16* b_b = a_b + OT_N;

    ot_init<<<16, 256, 0, stream>>>(cmax, b_f, b_b);
    ot_maxc<<<2048, 256, 0, stream>>>((const unsigned*)C, cmax, OT_N * OT_N);
    ot_buildK<<<dim3(OT_N / 32, OT_N / 32), dim3(32, 8), 0, stream>>>(C, cmax, K, KT);

    for (int it = 0; it < 100; ++it) {
        ot_matvec_div<<<OT_N / 16, MV_TPB, 0, stream>>>(K,  b_b, p, a_f, a_b);
        ot_matvec_div<<<OT_N / 16, MV_TPB, 0, stream>>>(KT, a_b, q, b_f, b_b);
    }

    ot_plan<<<2048, 256, 0, stream>>>(C, cmax, a_f, b_f, out);
}
